// MSPoolAttention_36069135352473
// MI455X (gfx1250) — matrix operand order repaired and checked by reading, not compiled, not run on hardware
//
#include <hip/hip_runtime.h>

// ---------------------------------------------------------------------------
// MSPoolAttention for MI455X (gfx1250, wave32, WMMA 16x16x32 f16 -> f32)
// B=2, C=256, H=W=48 (N=2304 tokens), 8 heads x d=32
// ---------------------------------------------------------------------------

typedef __attribute__((ext_vector_type(16))) _Float16 v16h;
typedef __attribute__((ext_vector_type(8)))  float    v8f;

#define T_TOK  2304
#define CDIM   256
#define KVDIM  512
#define HDIM   32
#define NHEAD  8
#define BATCH  2
#define SCALE_F 0.17677669529663687f   // 32^-0.5
#define TTILES (T_TOK / 16)            // 144

union AF {
    v16h   v;
    float4 f4[2];
    _Float16 h[16];
};

union PK8 {
    float4   f4;
    _Float16 h[8];
};

__device__ __forceinline__ float4 ldg4(const _Float16* p) {
    return *(const float4*)p;
}

__device__ __forceinline__ v8f wmma_f16(v16h a, v16h b, v8f c) {
    // D = A(16x32, f16) x B(32x16, f16) + C(16x16, f32)
    return __builtin_amdgcn_wmma_f32_16x16x32_f16(
        /*neg_a=*/false, a, /*neg_b=*/false, b,
        /*c_mod=*/(short)0, c, /*reuse_a=*/false, /*reuse_b=*/false);
}

// ---------------------------------------------------------------------------
// f32 -> f16 weight conversion (q_w / kv_w / proj_w are [O][C] row-major,
// which is exactly the B-operand friendly layout: per-lane contiguous K)
// ---------------------------------------------------------------------------
__global__ void cvt_f16_kernel(_Float16* __restrict__ dst,
                               const float* __restrict__ src, int n) {
    int i = blockIdx.x * 256 + threadIdx.x;
    if (i < n) dst[i] = (_Float16)src[i];
}

// ---------------------------------------------------------------------------
// Pool + transpose + f16 pre-pass.
// x: [B][C][48][48] f32.  Writes:
//   XT [b][n][c] f16  = x           (for Q projection)
//   XAT[b][n][c] f16  = (x + avg3 + avg5 + avg7)/4   (for KV projection)
// AvgPool: stride 1, count_include_pad (divide by k*k, OOB contributes 0).
// One block per (b,n), threadIdx.x = channel -> contiguous f16 stores.
// ---------------------------------------------------------------------------
__global__ void pool_transpose_kernel(const float* __restrict__ x,
                                      _Float16* __restrict__ XT,
                                      _Float16* __restrict__ XAT) {
    int n = blockIdx.x % T_TOK;
    int b = blockIdx.x / T_TOK;
    int c = threadIdx.x;
    int y  = n / 48;
    int xx = n % 48;
    const float* xc = x + ((size_t)(b * CDIM + c)) * T_TOK;

    float s3 = 0.f, s5 = 0.f, s7 = 0.f;
    #pragma unroll
    for (int dy = -3; dy <= 3; ++dy) {
        int yy = y + dy;
        if (yy < 0 || yy >= 48) continue;   // zero padding
        #pragma unroll
        for (int dx = -3; dx <= 3; ++dx) {
            int x2 = xx + dx;
            if (x2 < 0 || x2 >= 48) continue;
            float v = xc[yy * 48 + x2];
            s7 += v;
            if (dy >= -2 && dy <= 2 && dx >= -2 && dx <= 2) s5 += v;
            if (dy >= -1 && dy <= 1 && dx >= -1 && dx <= 1) s3 += v;
        }
    }
    float xv = xc[n];
    float av = (xv + s3 * (1.f / 9.f) + s5 * (1.f / 25.f) + s7 * (1.f / 49.f)) * 0.25f;

    size_t o = ((size_t)(b * T_TOK + n)) * CDIM + c;
    XT [o] = (_Float16)xv;
    XAT[o] = (_Float16)av;
}

// ---------------------------------------------------------------------------
// Generic projection GEMM:  out[b][t][o] (f16) = (sum_c A[b][t][c]*W[o][c] + bias[o]) * postScale
// A token-major/channel-contiguous (A operand), W row-major (B operand).
// One wave per 16x16 output tile.  K loop: 8 steps of 32 with one WMMA each.
// If VT != nullptr (KV case, O=512): channels o>=256 (the V half) are also
// scattered transposed into VT[b][o-256][t] with packed b128 stores.
// ---------------------------------------------------------------------------
__global__ void gemm_tn_kernel(const _Float16* __restrict__ A,
                               const _Float16* __restrict__ W,
                               const float* __restrict__ bias,
                               float postScale,
                               _Float16* __restrict__ out, int O,
                               _Float16* __restrict__ VT) {
    int tilesO = O / 16;
    int idx = blockIdx.x;
    int o0 = (idx % tilesO) * 16;  idx /= tilesO;
    int t0 = (idx % TTILES) * 16;  idx /= TTILES;
    int b  = idx;

    int lane = threadIdx.x;
    int H = lane >> 4;      // lane half
    int r = lane & 15;

    const _Float16* Ar = A + ((size_t)b * T_TOK + t0 + r) * CDIM;  // A row: token t0+r
    const _Float16* Wr = W + (size_t)(o0 + r) * CDIM;              // B col: output o0+r

    v8f acc = {};
    #pragma unroll
    for (int k0 = 0; k0 < CDIM; k0 += 32) {
        AF a, bw;
        // A lane layout (16x32 f16): K = [8H,8H+8) U [8H+16,8H+24)
        a.f4[0] = ldg4(Ar + k0 + 8 * H);
        a.f4[1] = ldg4(Ar + k0 + 8 * H + 16);
        // B lane layout (32x16 f16): K = [8H,8H+8) U [8H+16,8H+24)
        bw.f4[0] = ldg4(Wr + k0 + 8 * H);
        bw.f4[1] = ldg4(Wr + k0 + 8 * H + 16);
        acc = wmma_f16(a.v, bw.v, acc);
    }

    float bv = bias[o0 + r];
    float res[8];
    #pragma unroll
    for (int g = 0; g < 8; ++g) res[g] = (acc[g] + bv) * postScale;

    // D layout: lane (H,r) holds rows t = t0 + 8H + g, col o = o0 + r
    #pragma unroll
    for (int g = 0; g < 8; ++g) {
        int t = t0 + 8 * H + g;
        out[((size_t)b * T_TOK + t) * O + (o0 + r)] = (_Float16)res[g];
    }

    if (VT != nullptr && (o0 + r) >= CDIM) {
        // V half: also store transposed VT[oc][t], 8 consecutive t -> one b128
        PK8 pk;
        #pragma unroll
        for (int g = 0; g < 8; ++g) pk.h[g] = (_Float16)res[g];
        int oc = o0 + r - CDIM;                // = h*32 + dd
        *(float4*)(VT + ((size_t)b * CDIM + oc) * T_TOK + t0 + 8 * H) = pk.f4;
    }
}

// ---------------------------------------------------------------------------
// Flash attention, one wave per (b, head, 16-token tile).
//   S^T = K x Q^T   (so softmax axis m lives inside each lane: 8 VGPRs + one
//                    shfl_xor(16) — per-lane scalar running max/sum)
//   Y^T = V^T x P^T (V pre-transposed in VT; P^T rebuilt into B layout with
//                    8 shfl_xor(16) + selects per 32-key chunk)
// Q is pre-scaled by SCALE at projection time.
// Writes Y token-major [b][t][c] f16 (packed b128 stores) for the proj GEMM.
// ---------------------------------------------------------------------------
__global__ void attn_kernel(const _Float16* __restrict__ Q,
                            const _Float16* __restrict__ KVT,
                            const _Float16* __restrict__ VT,
                            _Float16* __restrict__ Y) {
    int idx = blockIdx.x;
    int tt = idx % TTILES;  idx /= TTILES;
    int h  = idx % NHEAD;
    int b  = idx / NHEAD;
    int t0 = tt * 16;

    int lane = threadIdx.x;
    int H = lane >> 4;
    int r = lane & 15;

    // Q^T as B operand: lane = query column n = t0+r, K = d in [8H,8H+8) U [8H+16,8H+24)
    AF qb;
    const _Float16* qrow = Q + ((size_t)b * T_TOK + t0 + r) * CDIM + h * HDIM + 8 * H;
    qb.f4[0] = ldg4(qrow);
    qb.f4[1] = ldg4(qrow + 16);

    const _Float16* kb = KVT + (size_t)b * T_TOK * KVDIM + h * HDIM;          // K rows [m][d]
    const _Float16* vb = VT  + ((size_t)b * CDIM + h * HDIM) * T_TOK;         // V^T rows [d][m]

    float Mx = -1e30f, L = 0.f;
    v8f y0 = {}, y1 = {};

    for (int m0 = 0; m0 < T_TOK; m0 += 32) {
        // --- S^T tiles: A = K[m0..+16)x d32 and K[m0+16..+32) ---
        AF ka0, ka1;
        const _Float16* k0p = kb + (size_t)(m0 + r) * KVDIM + 8 * H;
        ka0.f4[0] = ldg4(k0p);
        ka0.f4[1] = ldg4(k0p + 16);
        const _Float16* k1p = k0p + (size_t)16 * KVDIM;
        ka1.f4[0] = ldg4(k1p);
        ka1.f4[1] = ldg4(k1p + 16);

        v8f z = {};
        v8f s0 = wmma_f16(ka0.v, qb.v, z);   // rows m = m0 + 8H + g, col n = t0+r
        v8f s1 = wmma_f16(ka1.v, qb.v, z);   // rows m = m0 + 16 + 8H + g

        // --- online softmax stats (per query column n) ---
        float mx = -1e30f;
        #pragma unroll
        for (int g = 0; g < 8; ++g) { mx = fmaxf(mx, s0[g]); mx = fmaxf(mx, s1[g]); }
        mx = fmaxf(mx, __shfl_xor(mx, 16, 32));
        float newM  = fmaxf(Mx, mx);
        float alpha = __expf(Mx - newM);

        float p0[8], p1[8];
        float sum = 0.f;
        #pragma unroll
        for (int g = 0; g < 8; ++g) {
            p0[g] = __expf(s0[g] - newM);
            p1[g] = __expf(s1[g] - newM);
            sum += p0[g] + p1[g];
        }
        sum += __shfl_xor(sum, 16, 32);
        L  = L * alpha + sum;
        Mx = newM;
        #pragma unroll
        for (int g = 0; g < 8; ++g) { y0[g] = y0[g] * alpha; y1[g] = y1[g] * alpha; }

        // --- rebuild P^T into the B-operand layout ---
        // B element e (K = m_local = 16H + e): tile = H, srcHalf = (e>=8), g = e&7
        float nearv[8], farv[8];
        #pragma unroll
        for (int g = 0; g < 8; ++g) {
            float nr = H ? p1[g] : p0[g];     // tile_H[g] at own half
            float cr = H ? p0[g] : p1[g];     // partner sends tile_H[g] of other half
            nearv[g] = nr;
            farv[g]  = __shfl_xor(cr, 16, 32);
        }
        AF pb;
        #pragma unroll
        for (int e = 0; e < 16; ++e) {
            float v = (((e >> 3) & 1) == H) ? nearv[e & 7] : farv[e & 7];
            pb.h[e] = (_Float16)v;
        }

        // --- Y^T += V^T x P^T (two d-tiles: d 0..15 and 16..31) ---
        AF va0, va1;
        const _Float16* v0p = vb + (size_t)r * T_TOK + m0 + 16 * H;
        va0.f4[0] = ldg4(v0p);
        va0.f4[1] = ldg4(v0p + 8);
        const _Float16* v1p = v0p + (size_t)16 * T_TOK;
        va1.f4[0] = ldg4(v1p);
        va1.f4[1] = ldg4(v1p + 8);

        y0 = wmma_f16(va0.v, pb.v, y0);
        y1 = wmma_f16(va1.v, pb.v, y1);
    }

    float inv = 1.f / L;
    PK8 pk0, pk1;
    #pragma unroll
    for (int g = 0; g < 8; ++g) {
        pk0.h[g] = (_Float16)(y0[g] * inv);   // d = 8H + g
        pk1.h[g] = (_Float16)(y1[g] * inv);   // d = 16 + 8H + g
    }
    _Float16* yrow = Y + ((size_t)b * T_TOK + t0 + r) * CDIM + h * HDIM;
    *(float4*)(yrow + 8 * H)      = pk0.f4;
    *(float4*)(yrow + 16 + 8 * H) = pk1.f4;
}

// ---------------------------------------------------------------------------
// Final projection GEMM, f32 output directly in [B][C][48][48] order.
// out[b][o][t] = sum_c Y[b][t][c]*proj_w[o][c] + proj_b[o]
// Per lane: fixed o, 8 consecutive t -> two float4 stores.
// ---------------------------------------------------------------------------
__global__ void gemm_proj_kernel(const _Float16* __restrict__ A,
                                 const _Float16* __restrict__ W,
                                 const float* __restrict__ bias,
                                 float* __restrict__ out) {
    int idx = blockIdx.x;
    int o0 = (idx % 16) * 16;      idx /= 16;
    int t0 = (idx % TTILES) * 16;  idx /= TTILES;
    int b  = idx;

    int lane = threadIdx.x;
    int H = lane >> 4;
    int r = lane & 15;

    const _Float16* Ar = A + ((size_t)b * T_TOK + t0 + r) * CDIM;
    const _Float16* Wr = W + (size_t)(o0 + r) * CDIM;

    v8f acc = {};
    #pragma unroll
    for (int k0 = 0; k0 < CDIM; k0 += 32) {
        AF a, bw;
        a.f4[0]  = ldg4(Ar + k0 + 8 * H);
        a.f4[1]  = ldg4(Ar + k0 + 8 * H + 16);
        bw.f4[0] = ldg4(Wr + k0 + 8 * H);
        bw.f4[1] = ldg4(Wr + k0 + 8 * H + 16);
        acc = wmma_f16(a.v, bw.v, acc);
    }

    float bv = bias[o0 + r];
    float res[8];
    #pragma unroll
    for (int g = 0; g < 8; ++g) res[g] = acc[g] + bv;

    float* orow = out + ((size_t)b * CDIM + (o0 + r)) * T_TOK + t0 + 8 * H;
    *(float4*)(orow)     = make_float4(res[0], res[1], res[2], res[3]);
    *(float4*)(orow + 4) = make_float4(res[4], res[5], res[6], res[7]);
}

// ---------------------------------------------------------------------------
// Launch
// ---------------------------------------------------------------------------
extern "C" void kernel_launch(void* const* d_in, const int* in_sizes, int n_in,
                              void* d_out, int out_size, void* d_ws, size_t ws_size,
                              hipStream_t stream) {
    (void)in_sizes; (void)n_in; (void)out_size; (void)ws_size;

    const float* x      = (const float*)d_in[0];
    const float* q_w    = (const float*)d_in[1];
    const float* q_b    = (const float*)d_in[2];
    const float* kv_w   = (const float*)d_in[3];
    const float* kv_b   = (const float*)d_in[4];
    const float* proj_w = (const float*)d_in[5];
    const float* proj_b = (const float*)d_in[6];
    float*       out    = (float*)d_out;

    // workspace partition (all 256B aligned)
    uint8_t* ws = (uint8_t*)d_ws;
    size_t off = 0;
    auto alloc = [&](size_t bytes) {
        uint8_t* p = ws + off;
        off += (bytes + 255) & ~(size_t)255;
        return p;
    };
    const size_t szTC  = (size_t)BATCH * T_TOK * CDIM  * sizeof(_Float16); // 2.36 MB
    const size_t szTKV = (size_t)BATCH * T_TOK * KVDIM * sizeof(_Float16); // 4.72 MB

    _Float16* XT   = (_Float16*)alloc(szTC);                       // x, token-major
    _Float16* XAT  = (_Float16*)alloc(szTC);                       // pooled avg, token-major
    _Float16* Qb   = (_Float16*)alloc(szTC);                       // Q*SCALE, [b][t][256]
    _Float16* KVTb = (_Float16*)alloc(szTKV);                      // [b][t][512] (k:0..255, v:256..511)
    _Float16* VTb  = (_Float16*)alloc(szTC);                       // V transposed [b][oc][t]
    _Float16* Yb   = (_Float16*)alloc(szTC);                       // attention out, [b][t][256]
    _Float16* Wq   = (_Float16*)alloc((size_t)CDIM  * CDIM * 2);
    _Float16* Wkv  = (_Float16*)alloc((size_t)KVDIM * CDIM * 2);
    _Float16* Wp   = (_Float16*)alloc((size_t)CDIM  * CDIM * 2);

    // 1) weights -> f16
    cvt_f16_kernel<<<(CDIM * CDIM + 255) / 256, 256, 0, stream>>>(Wq, q_w, CDIM * CDIM);
    cvt_f16_kernel<<<(KVDIM * CDIM + 255) / 256, 256, 0, stream>>>(Wkv, kv_w, KVDIM * CDIM);
    cvt_f16_kernel<<<(CDIM * CDIM + 255) / 256, 256, 0, stream>>>(Wp, proj_w, CDIM * CDIM);

    // 2) pool + transpose + f16
    pool_transpose_kernel<<<BATCH * T_TOK, CDIM, 0, stream>>>(x, XT, XAT);

    // 3) Q projection (SCALE folded in) and KV projection (+ V transpose)
    gemm_tn_kernel<<<BATCH * TTILES * (CDIM / 16), 32, 0, stream>>>(
        XT, Wq, q_b, SCALE_F, Qb, CDIM, nullptr);
    gemm_tn_kernel<<<BATCH * TTILES * (KVDIM / 16), 32, 0, stream>>>(
        XAT, Wkv, kv_b, 1.0f, KVTb, KVDIM, VTb);

    // 4) flash attention
    attn_kernel<<<BATCH * NHEAD * TTILES, 32, 0, stream>>>(Qb, KVTb, VTb, Yb);

    // 5) output projection, f32 channel-major
    gemm_proj_kernel<<<BATCH * TTILES * (CDIM / 16), 32, 0, stream>>>(Yb, Wp, proj_b, out);
}
